// LSTMnet_2714419331322
// MI455X (gfx1250) — compile-verified, matrix operand order checked by reading, not run on hardware
//
#include <hip/hip_runtime.h>

#define BATCH   128
#define SEQ     512
#define IN_DIM  1024
#define HID     1024
#define OUT_DIM 512

#if defined(__gfx1250__) && __has_builtin(__builtin_amdgcn_global_load_async_to_lds_b128) && \
    __has_builtin(__builtin_amdgcn_s_wait_asynccnt)
#define USE_ASYNC_LDS 1
#else
#define USE_ASYNC_LDS 0
#endif

typedef __attribute__((ext_vector_type(16))) __bf16 v16bf;
typedef __attribute__((ext_vector_type(8)))  __bf16 v8bf;
typedef __attribute__((ext_vector_type(8)))  float  v8f;
typedef __attribute__((ext_vector_type(4)))  float  v4f;
typedef int b128_t __attribute__((vector_size(16)));   // matches builtin param type

// fp32 -> bf16 round-to-nearest-even (bit manipulation only; no __bf16 arithmetic)
__device__ __forceinline__ __bf16 f2bf(float f) {
    union { float f; unsigned int u; } in; in.f = f;
    unsigned int u = in.u;
    u += 0x7fffu + ((u >> 16) & 1u);
    union { unsigned short s; __bf16 b; } out;
    out.s = (unsigned short)(u >> 16);
    return out.b;
}

__device__ __forceinline__ float sigmoidf_(float x) {
    return 1.0f / (1.0f + __expf(-x));
}

// Issue a 16B async copy global -> LDS (per-lane addresses), tracked by ASYNCcnt.
__device__ __forceinline__ void async_cp16(void* lds, const void* g) {
#if USE_ASYNC_LDS
    __builtin_amdgcn_global_load_async_to_lds_b128(
        (__attribute__((address_space(1))) b128_t*)(g),
        (__attribute__((address_space(3))) b128_t*)(lds),
        0, 0);
#else
    *(v8bf*)lds = *(const v8bf*)g;   // fallback: load + ds_store
#endif
}

__device__ __forceinline__ void async_cp_wait() {
#if USE_ASYNC_LDS
    __builtin_amdgcn_s_wait_asynccnt(0);
#endif
}

// Load a 16x32 bf16 tile (rows row0..row0+15, cols k0..k0+31) from a row-major
// matrix with row stride ld into the WMMA A/B VGPR layout (ISA 7.12.2):
//   lanes 0-15 : row = lane,    elems 0-7 -> K=k0+0..7,  elems 8-15 -> K=k0+16..23
//   lanes 16-31: row = lane-16, elems 0-7 -> K=k0+8..15, elems 8-15 -> K=k0+24..31
__device__ __forceinline__ v16bf load_tile_bf(const __bf16* base, size_t ld,
                                              int row0, int k0, int lane) {
    int r  = row0 + (lane & 15);
    int kb = (lane >> 4) << 3;            // 0 or 8
    const __bf16* p = base + (size_t)r * ld + (size_t)(k0 + kb);
    v8bf lo = *(const v8bf*)(p);          // 16B load
    v8bf hi = *(const v8bf*)(p + 16);     // 16B load
    v16bf out;
#pragma unroll
    for (int i = 0; i < 8; ++i) { out[i] = lo[i]; out[i + 8] = hi[i]; }
    return out;
}

// Same tile layout, but source matrix is fp32 (converted on the fly).
__device__ __forceinline__ v16bf load_tile_f32(const float* base, size_t ld,
                                               int row0, int k0, int lane) {
    int r  = row0 + (lane & 15);
    int kb = (lane >> 4) << 3;
    const float* p = base + (size_t)r * ld + (size_t)(k0 + kb);
    v4f a0 = *(const v4f*)(p);
    v4f a1 = *(const v4f*)(p + 4);
    v4f a2 = *(const v4f*)(p + 16);
    v4f a3 = *(const v4f*)(p + 20);
    v16bf out;
#pragma unroll
    for (int i = 0; i < 4; ++i) {
        out[i]      = f2bf(a0[i]);
        out[i + 4]  = f2bf(a1[i]);
        out[i + 8]  = f2bf(a2[i]);
        out[i + 12] = f2bf(a3[i]);
    }
    return out;
}

__device__ __forceinline__ v8f wmma_bf16(v16bf a, v16bf b, v8f c) {
    return __builtin_amdgcn_wmma_f32_16x16x32_bf16(false, a, false, b,
                                                   (short)0, c, false, false);
}

// Rebuild a B fragment from its LDS slot ([piece][lane], 16B per lane per piece)
__device__ __forceinline__ v16bf frag_from_lds(v8bf (&slot)[2][32], int lane) {
    v8bf lo = slot[0][lane];   // ds_load_b128
    v8bf hi = slot[1][lane];   // ds_load_b128
    v16bf out;
#pragma unroll
    for (int i = 0; i < 8; ++i) { out[i] = lo[i]; out[i + 8] = hi[i]; }
    return out;
}

// One GEMM phase of a GRU step: acc_{r,z,n} += x_tile @ W[{r,z,n} rows]^T.
// All 8 waves of the block share n0, so the 3 B fragments per k-step are staged
// in LDS once per block by waves 0-2 (async global->LDS copies, double-buffered,
// one barrier per k-step; ASYNCcnt drained before the barrier).
template <bool XF32>
__device__ __forceinline__ void gemm_phase(
    const void* __restrict__ xbase, size_t x_ld, int Dk,
    const __bf16* __restrict__ W, size_t w_ld,
    int m0, int n0, int lane, int wave,
    v8bf (&smem)[2][3][2][32],
    v8f& ar, v8f& az, v8f& an)
{
    const int H = HID;
    int kb = (lane >> 4) << 3;
    const __bf16* wrow = nullptr;
    if (wave < 3) {
        // loader wave g handles gate g's 16 weight rows at n0 + g*H
        int r = n0 + wave * H + (lane & 15);
        wrow = W + (size_t)r * w_ld + (size_t)kb;
        async_cp16(&smem[0][wave][0][lane], wrow);
        async_cp16(&smem[0][wave][1][lane], wrow + 16);
        async_cp_wait();
    }
    __syncthreads();

    int p = 0;
    for (int k0 = 0; k0 < Dk; k0 += 32) {
        // loaders: kick off next k-block's async copies early so L2 latency
        // overlaps this iteration's ds_load + 3x WMMA
        const bool have_next = (k0 + 32) < Dk;
        if (wave < 3 && have_next) {
            async_cp16(&smem[p ^ 1][wave][0][lane], wrow + (k0 + 32));
            async_cp16(&smem[p ^ 1][wave][1][lane], wrow + (k0 + 48));
            __builtin_prefetch((const void*)(wrow + k0 + 96), 0, 1);
        }

        v16bf a;
        if constexpr (XF32) a = load_tile_f32((const float*)xbase, x_ld, m0, k0, lane);
        else                a = load_tile_bf((const __bf16*)xbase, x_ld, m0, k0, lane);

        v16bf b0 = frag_from_lds(smem[p][0], lane);
        v16bf b1 = frag_from_lds(smem[p][1], lane);
        v16bf b2 = frag_from_lds(smem[p][2], lane);

        ar = wmma_bf16(a, b0, ar);
        az = wmma_bf16(a, b1, az);
        an = wmma_bf16(a, b2, an);

        if (wave < 3) async_cp_wait();   // copies for buf p^1 have landed
        __syncthreads();
        p ^= 1;
    }
}

// One GRU cell step: h_new = GRUCell(x, h_old). One wave per 16x16 tile of h.
// Grid: 64 blocks x 256 threads (8 waves) = 512 waves = (B/16) x (H/16) tiles.
// Block b owns column tile n0 = b*16 for all 8 batch tiles (waves).
template <bool XF32>
__global__ void __launch_bounds__(256) gru_step_kernel(
    const void* __restrict__ xbase, size_t x_ld, int Dk,
    const __bf16* __restrict__ Wih, const __bf16* __restrict__ Whh,
    const float* __restrict__ bih, const float* __restrict__ bhh,
    const float* __restrict__ hold_f, const __bf16* __restrict__ hold_bf,
    float* __restrict__ hnew_f, __bf16* __restrict__ hnew_bf)
{
    const int H = HID;
    __shared__ v8bf smem[2][3][2][32];   // [buf][gate][piece][lane] = 6 KB

    int lane = threadIdx.x & 31;
    int wave = threadIdx.x >> 5;
    int tile = blockIdx.x * 8 + wave;
    int m0 = (tile & 7) << 4;      // batch tile (0..127)
    int n0 = (tile >> 3) << 4;     // hidden tile (0..1023), uniform per block

    v8f a_ir = {}, a_iz = {}, a_in = {};
    v8f a_hr = {}, a_hz = {}, a_hn = {};

    // gi = x @ W_ih^T
    gemm_phase<XF32>(xbase, x_ld, Dk, Wih, (size_t)Dk,
                     m0, n0, lane, wave, smem, a_ir, a_iz, a_in);
    // gh = h_old @ W_hh^T
    gemm_phase<false>((const void*)hold_bf, (size_t)H, H, Whh, (size_t)H,
                      m0, n0, lane, wave, smem, a_hr, a_hz, a_hn);

    // Gate math on the C/D layout: lane -> N = lane&15 (+n0), VGPR i -> M = i + 8*(lane>=16)
    int nn  = n0 + (lane & 15);
    int mh  = (lane >> 4) << 3;
    float b_r  = bih[nn]         + bhh[nn];
    float b_z  = bih[H + nn]     + bhh[H + nn];
    float b_in = bih[2 * H + nn];
    float b_hn = bhh[2 * H + nn];
#pragma unroll
    for (int i = 0; i < 8; ++i) {
        int m = m0 + mh + i;
        float r  = sigmoidf_(a_ir[i] + a_hr[i] + b_r);
        float z  = sigmoidf_(a_iz[i] + a_hz[i] + b_z);
        float ng = tanhf(a_in[i] + b_in + r * (a_hn[i] + b_hn));
        float hp = hold_f[(size_t)m * H + nn];
        float h  = (1.0f - z) * ng + z * hp;
        hnew_f[(size_t)m * H + nn]  = h;
        hnew_bf[(size_t)m * H + nn] = f2bf(h);
    }
}

// out = h1 @ W_fc^T + b_fc : (B/16) x (OUT/16) = 8 x 32 tiles -> 32 blocks x 8 waves
__global__ void __launch_bounds__(256) fc_kernel(
    const __bf16* __restrict__ h, const __bf16* __restrict__ W,
    const float* __restrict__ b, float* __restrict__ out)
{
    int lane = threadIdx.x & 31;
    int wave = threadIdx.x >> 5;
    int tile = blockIdx.x * 8 + wave;
    int m0 = (tile & 7) << 4;
    int n0 = (tile >> 3) << 4;
    v8f acc = {};
    for (int k0 = 0; k0 < HID; k0 += 32) {
        v16bf a  = load_tile_bf(h, (size_t)HID, m0, k0, lane);
        v16bf bb = load_tile_bf(W, (size_t)HID, n0, k0, lane);
        acc = wmma_bf16(a, bb, acc);
    }
    int nn = n0 + (lane & 15);
    int mh = (lane >> 4) << 3;
    float bias = b[nn];
#pragma unroll
    for (int i = 0; i < 8; ++i) {
        int m = m0 + mh + i;
        out[(size_t)m * OUT_DIM + nn] = acc[i] + bias;
    }
}

__global__ void cvt_kernel(const float* __restrict__ src, __bf16* __restrict__ dst, int n) {
    int i = blockIdx.x * blockDim.x + threadIdx.x;
    if (i < n) dst[i] = f2bf(src[i]);
}

__global__ void zero_kernel(unsigned int* __restrict__ p, int nwords) {
    int i = blockIdx.x * blockDim.x + threadIdx.x;
    if (i < nwords) p[i] = 0u;
}

extern "C" void kernel_launch(void* const* d_in, const int* in_sizes, int n_in,
                              void* d_out, int out_size, void* d_ws, size_t ws_size,
                              hipStream_t stream) {
    const float* input = (const float*)d_in[0];
    const float* Wih0f = (const float*)d_in[1];
    const float* Whh0f = (const float*)d_in[2];
    const float* bih0  = (const float*)d_in[3];
    const float* bhh0  = (const float*)d_in[4];
    const float* Wih1f = (const float*)d_in[5];
    const float* Whh1f = (const float*)d_in[6];
    const float* bih1  = (const float*)d_in[7];
    const float* bhh1  = (const float*)d_in[8];
    const float* Wfcf  = (const float*)d_in[9];
    const float* bfc   = (const float*)d_in[10];
    float* out = (float*)d_out;

    // Workspace carve-up (256B aligned), ~29 MB total
    char* ws = (char*)d_ws;
    size_t off = 0;
    auto carve = [&](size_t bytes) -> void* {
        void* p = (void*)(ws + off);
        off = (off + bytes + 255) & ~(size_t)255;
        return p;
    };
    const int W0N = 3 * HID * IN_DIM;   // 3,145,728
    const int W1N = 3 * HID * HID;      // 3,145,728
    const int WFN = OUT_DIM * HID;      // 524,288
    __bf16* Wih0 = (__bf16*)carve((size_t)W0N * 2);
    __bf16* Whh0 = (__bf16*)carve((size_t)W1N * 2);
    __bf16* Wih1 = (__bf16*)carve((size_t)W1N * 2);
    __bf16* Whh1 = (__bf16*)carve((size_t)W1N * 2);
    __bf16* Wfc  = (__bf16*)carve((size_t)WFN * 2);

    const size_t hE = (size_t)BATCH * HID;         // 131072 elements
    float*  hf = (float*) carve(4 * hE * sizeof(float));   // h0[2], h1[2] fp32
    __bf16* hb = (__bf16*)carve(4 * hE * sizeof(__bf16));  // h0[2], h1[2] bf16
    float*  h0f[2] = { hf,          hf + hE };
    float*  h1f[2] = { hf + 2 * hE, hf + 3 * hE };
    __bf16* h0b[2] = { hb,          hb + hE };
    __bf16* h1b[2] = { hb + 2 * hE, hb + 3 * hE };

    // Pre-pass: weight conversion + zero h state (hf and hb are contiguous)
    cvt_kernel<<<(W0N + 255) / 256, 256, 0, stream>>>(Wih0f, Wih0, W0N);
    cvt_kernel<<<(W1N + 255) / 256, 256, 0, stream>>>(Whh0f, Whh0, W1N);
    cvt_kernel<<<(W1N + 255) / 256, 256, 0, stream>>>(Wih1f, Wih1, W1N);
    cvt_kernel<<<(W1N + 255) / 256, 256, 0, stream>>>(Whh1f, Whh1, W1N);
    cvt_kernel<<<(WFN + 255) / 256, 256, 0, stream>>>(Wfcf,  Wfc,  WFN);
    const int zwords = (int)((4 * hE * 4 + 4 * hE * 2) / 4);   // 786432
    zero_kernel<<<(zwords + 255) / 256, 256, 0, stream>>>((unsigned int*)hf, zwords);

    // Sequential scan over time; graph capture amortizes the launch chain.
    int c0 = 0, c1 = 0;
    for (int t = 0; t < SEQ; ++t) {
        // layer 0: x_t is input[:, t, :] (row stride SEQ*IN_DIM), fp32 -> bf16 inline
        gru_step_kernel<true><<<64, 256, 0, stream>>>(
            (const void*)(input + (size_t)t * IN_DIM), (size_t)SEQ * IN_DIM, IN_DIM,
            Wih0, Whh0, bih0, bhh0,
            h0f[c0], h0b[c0], h0f[c0 ^ 1], h0b[c0 ^ 1]);
        c0 ^= 1;
        // layer 1: x = h0_new (bf16)
        gru_step_kernel<false><<<64, 256, 0, stream>>>(
            (const void*)h0b[c0], (size_t)HID, HID,
            Wih1, Whh1, bih1, bhh1,
            h1f[c1], h1b[c1], h1f[c1 ^ 1], h1b[c1 ^ 1]);
        c1 ^= 1;
    }

    fc_kernel<<<32, 256, 0, stream>>>(h1b[c1], Wfc, bfc, out);
}
